// Model_53154515255439
// MI455X (gfx1250) — matrix operand order repaired and checked by reading, not compiled, not run on hardware
//
#include <hip/hip_runtime.h>
#include <math.h>

// ---------------- model constants ----------------
#define BB 16
#define SEQ 96
#define NVARS 862
#define NMARK 4
#define LTOK 866            // NVARS + NMARK (mamba sequence length)
#define DM 512
#define DSTATE 16
#define DTRANK 32
#define MROWS (BB * LTOK)   // 13856 = 866*16, divisible by 32
#define PRED 96

typedef __attribute__((ext_vector_type(16))) __bf16 v16bf;
typedef __attribute__((ext_vector_type(8)))  float  v8f;
typedef unsigned short ushort_t;

__device__ __forceinline__ ushort_t f2bf(float f) {
  union { float f; unsigned u; } c; c.f = f;
  unsigned r = c.u + 0x7FFFu + ((c.u >> 16) & 1u);   // round-to-nearest-even
  return (ushort_t)(r >> 16);
}

union Frag { v16bf v; uint4 q[2]; };

// ---------------------------------------------------------------------------
// WMMA GEMM on pre-converted bf16 operands.
//   outf/outb (M,N) = act( A(M,K)bf16 * W(N,K)bf16^T + bias )
// One wave per block; wave computes a (16*MT) x (16*NT) tile, K stepped by 32.
// Fragments are assembled with raw b128 loads (no conversion VALU in loop):
//   A lane frag = 16B chunks at K = kb+8*hi and kb+16+8*hi
//   B lane frag = 16B chunks at K = kb+8*hi and kb+16+8*hi
// ACT: 0=none 1=relu 2=softplus.  OUTF bit0: f32 store, bit1: bf16 store.
// ---------------------------------------------------------------------------
template <int MT, int NT, int ACT, int OUTF>
__global__ __launch_bounds__(32) void gemm_wmma_kernel(
    const ushort_t* __restrict__ A, int lda,
    const ushort_t* __restrict__ W, int ldw,
    const float* __restrict__ bias,
    float* __restrict__ outf, ushort_t* __restrict__ outb,
    int ldo, int K)
{
  const int lane = threadIdx.x & 31;
  const int lo   = lane & 15;
  const int hi   = lane >> 4;
  const int rowBase = blockIdx.x * 16 * MT;
  const int colBase = blockIdx.y * 16 * NT;

  v8f acc[MT][NT] = {};

  for (int kb = 0; kb < K; kb += 32) {
    Frag a[MT];
#pragma unroll
    for (int m = 0; m < MT; ++m) {
      const ushort_t* ap = A + (size_t)(rowBase + 16 * m + lo) * lda + kb + 8 * hi;
      a[m].q[0] = ((const uint4*)ap)[0];   // K: kb+8hi .. +7
      a[m].q[1] = ((const uint4*)ap)[2];   // K: kb+16+8hi .. +7
    }
#pragma unroll
    for (int t = 0; t < NT; ++t) {
      const ushort_t* bp = W + (size_t)(colBase + 16 * t + lo) * ldw + kb + 8 * hi;
      Frag b;
      b.q[0] = ((const uint4*)bp)[0];      // K: kb+8hi .. +7
      b.q[1] = ((const uint4*)bp)[2];      // K: kb+16+8hi .. +7
#pragma unroll
      for (int m = 0; m < MT; ++m)
        acc[m][t] = __builtin_amdgcn_wmma_f32_16x16x32_bf16(
            false, a[m].v, false, b.v, (short)0, acc[m][t], false, false);
    }
  }

  // C/D layout: VGPR r -> M = r + 8*hi, N = lane&15
#pragma unroll
  for (int t = 0; t < NT; ++t) {
    int col = colBase + 16 * t + lo;
    float bb = bias ? bias[col] : 0.f;
#pragma unroll
    for (int m = 0; m < MT; ++m) {
#pragma unroll
      for (int r = 0; r < 8; ++r) {
        int orow = rowBase + 16 * m + r + 8 * hi;
        float v = acc[m][t][r] + bb;
        if (ACT == 1) v = fmaxf(v, 0.f);
        else if (ACT == 2) v = (v > 20.f) ? v : log1pf(__expf(v));
        if (OUTF & 1) outf[(size_t)orow * ldo + col] = v;
        if (OUTF & 2) outb[(size_t)orow * ldo + col] = f2bf(v);
      }
    }
  }
}

// generic fp32 -> bf16 tensor conversion (weights, once per call)
__global__ void cvt_bf16_kernel(const float* __restrict__ in,
                                ushort_t* __restrict__ out, size_t n)
{
  size_t i = (size_t)blockIdx.x * blockDim.x + threadIdx.x;
  if (i < n) out[i] = f2bf(in[i]);
}

// ---------------------------------------------------------------------------
// RevIN prep: per (b, var) mean/std over SEQ; build bf16 tok (B, LTOK, SEQ)
// ---------------------------------------------------------------------------
__global__ void prep_kernel(const float* __restrict__ xenc,
                            const float* __restrict__ xmark,
                            ushort_t* __restrict__ tokb,
                            float* __restrict__ meanb,
                            float* __restrict__ stdb)
{
  int idx = blockIdx.x * blockDim.x + threadIdx.x;
  if (idx >= BB * LTOK) return;
  int b = idx / LTOK, n = idx % LTOK;
  ushort_t* trow = tokb + (size_t)idx * SEQ;
  if (n < NVARS) {
    const float* src = xenc + (size_t)b * SEQ * NVARS + n;
    float s = 0.f;
    for (int t = 0; t < SEQ; ++t) s += src[(size_t)t * NVARS];
    float mu = s / SEQ;
    float vs = 0.f;
    for (int t = 0; t < SEQ; ++t) { float d = src[(size_t)t * NVARS] - mu; vs += d * d; }
    float sd = sqrtf(vs / SEQ + 1e-5f);
    meanb[b * NVARS + n] = mu;
    stdb[b * NVARS + n]  = sd;
    float inv = 1.f / sd;
    for (int t = 0; t < SEQ; ++t) trow[t] = f2bf((src[(size_t)t * NVARS] - mu) * inv);
  } else {
    const float* src = xmark + (size_t)b * SEQ * NMARK + (n - NVARS);
    for (int t = 0; t < SEQ; ++t) trow[t] = f2bf(src[(size_t)t * NMARK]);
  }
}

// flip token axis on bf16 activations (only GEMMs consume the reversed seq)
__global__ void flip_bf_kernel(const ushort_t* __restrict__ in,
                               ushort_t* __restrict__ out)
{
  size_t i = (size_t)blockIdx.x * blockDim.x + threadIdx.x;
  if (i >= (size_t)MROWS * DM) return;
  int d = (int)(i % DM);
  size_t r = i / DM;
  int l = (int)(r % LTOK);
  int b = (int)(r / LTOK);
  out[((size_t)b * LTOK + (LTOK - 1 - l)) * DM + d] = in[i];
}

// depthwise width-2 causal conv + SiLU; emits f32 (scan) + bf16 (xproj GEMM)
__global__ void conv_silu_kernel(const float* __restrict__ xz,
                                 const float* __restrict__ cw,
                                 const float* __restrict__ cb,
                                 float* __restrict__ xc,
                                 ushort_t* __restrict__ xcb)
{
  size_t i = (size_t)blockIdx.x * blockDim.x + threadIdx.x;
  if (i >= (size_t)MROWS * DM) return;
  int d = (int)(i % DM);
  size_t r = i / DM;
  int l = (int)(r % LTOK);
  float prev = (l > 0) ? xz[(r - 1) * (2 * DM) + d] : 0.f;
  float cur  = xz[r * (2 * DM) + d];
  float v = prev * cw[2 * d] + cur * cw[2 * d + 1] + cb[d];
  float sv = v / (1.f + __expf(-v));
  xc[i]  = sv;
  xcb[i] = f2bf(sv);
}

// ---------------------------------------------------------------------------
// Selective scan: lane per (b, d, s). 866 sequential steps; width-16 shfl
// reduction over the state dim each step. (fp32 for recurrence accuracy)
// ---------------------------------------------------------------------------
__global__ __launch_bounds__(256) void scan_kernel(
    const float* __restrict__ xc, const float* __restrict__ dt,
    const float* __restrict__ dbc, const float* __restrict__ A_log,
    const float* __restrict__ Dp, float* __restrict__ y)
{
  int idx = blockIdx.x * 256 + threadIdx.x;     // 16*512*16 = 131072 threads
  int s = idx & 15;
  int p = idx >> 4;
  int b = p >> 9;
  int d = p & (DM - 1);
  float a  = -__expf(A_log[d * DSTATE + s]);
  float Dd = Dp[d];
  const float* dtp = dt  + (size_t)b * LTOK * DM + d;
  const float* xp  = xc  + (size_t)b * LTOK * DM + d;
  const float* bcp = dbc + (size_t)b * LTOK * 64;
  float* yp        = y   + (size_t)b * LTOK * DM + d;
  float h = 0.f;
  for (int t = 0; t < LTOK; ++t) {
    float dtv = dtp[(size_t)t * DM];
    float xv  = xp[(size_t)t * DM];
    float Bv  = bcp[t * 64 + DTRANK + s];
    float Cv  = bcp[t * 64 + DTRANK + DSTATE + s];
    h = __expf(dtv * a) * h + dtv * xv * Bv;
    float part = h * Cv;
    part += __shfl_xor(part, 8, 16);
    part += __shfl_xor(part, 4, 16);
    part += __shfl_xor(part, 2, 16);
    part += __shfl_xor(part, 1, 16);
    if (s == 0) yp[(size_t)t * DM] = part + xv * Dd;
  }
}

// g = y * silu(z)  -> bf16 (only consumer is out_proj GEMM)
__global__ void gate_kernel(const float* __restrict__ y,
                            const float* __restrict__ xz,
                            ushort_t* __restrict__ gb)
{
  size_t i = (size_t)blockIdx.x * blockDim.x + threadIdx.x;
  if (i >= (size_t)MROWS * DM) return;
  int d = (int)(i % DM);
  size_t r = i / DM;
  float z = xz[r * (2 * DM) + DM + d];
  gb[i] = f2bf(y[i] * (z / (1.f + __expf(-z))));
}

// xsum = x + fwd
__global__ void add_kernel(const float* __restrict__ x,
                           const float* __restrict__ f,
                           float* __restrict__ o)
{
  size_t i = (size_t)blockIdx.x * blockDim.x + threadIdx.x;
  if (i >= (size_t)MROWS * DM) return;
  o[i] = x[i] + f[i];
}

// x = xsum + flip(rev)
__global__ void combine_kernel(const float* __restrict__ xsum,
                               const float* __restrict__ rev,
                               float* __restrict__ x)
{
  size_t i = (size_t)blockIdx.x * blockDim.x + threadIdx.x;
  if (i >= (size_t)MROWS * DM) return;
  int d = (int)(i % DM);
  size_t r = i / DM;
  int l = (int)(r % LTOK);
  int b = (int)(r / LTOK);
  x[i] = xsum[i] + rev[((size_t)b * LTOK + (LTOK - 1 - l)) * DM + d];
}

// wave-per-row LayerNorm over DM=512; optional residual, f32 and/or bf16 out
__global__ __launch_bounds__(256) void ln_kernel(
    const float* __restrict__ in, const float* __restrict__ res,
    const float* __restrict__ w, const float* __restrict__ b,
    float* __restrict__ outf, ushort_t* __restrict__ outb)
{
  int wid = threadIdx.x >> 5;
  int lane = threadIdx.x & 31;
  size_t row = (size_t)blockIdx.x * 8 + wid;
  const float* ip = in + row * DM;
  const float* rp = res ? res + row * DM : nullptr;
  float v[16];
  float s = 0.f;
#pragma unroll
  for (int j = 0; j < 16; ++j) {
    float t = ip[lane + 32 * j];
    if (rp) t += rp[lane + 32 * j];
    v[j] = t; s += t;
  }
#pragma unroll
  for (int o = 16; o >= 1; o >>= 1) s += __shfl_xor(s, o, 32);
  float mu = s * (1.f / DM);
  float vs = 0.f;
#pragma unroll
  for (int j = 0; j < 16; ++j) { float d = v[j] - mu; vs += d * d; }
#pragma unroll
  for (int o = 16; o >= 1; o >>= 1) vs += __shfl_xor(vs, o, 32);
  float inv = rsqrtf(vs * (1.f / DM) + 1e-5f);
#pragma unroll
  for (int j = 0; j < 16; ++j) {
    int e = lane + 32 * j;
    float r2 = (v[j] - mu) * inv * w[e] + b[e];
    if (outf) outf[row * DM + e] = r2;
    if (outb) outb[row * DM + e] = f2bf(r2);
  }
}

// out[b,t,n] = dectmp[b,n,t] * std[b,n] + mean[b,n]  (n < NVARS)
__global__ void denorm_kernel(const float* __restrict__ dectmp,
                              const float* __restrict__ meanb,
                              const float* __restrict__ stdb,
                              float* __restrict__ out)
{
  size_t i = (size_t)blockIdx.x * blockDim.x + threadIdx.x;
  if (i >= (size_t)BB * PRED * NVARS) return;
  int n = (int)(i % NVARS);
  size_t r = i / NVARS;
  int t = (int)(r % PRED);
  int b = (int)(r / PRED);
  float v = dectmp[((size_t)b * LTOK + n) * PRED + t];
  out[i] = v * stdb[b * NVARS + n] + meanb[b * NVARS + n];
}

// ---------------------------------------------------------------------------
static inline int nb(size_t n) { return (int)((n + 255) / 256); }

extern "C" void kernel_launch(void* const* d_in, const int* in_sizes, int n_in,
                              void* d_out, int out_size, void* d_ws, size_t ws_size,
                              hipStream_t stream)
{
  (void)in_sizes; (void)n_in; (void)out_size; (void)ws_size;

  const float* x_enc  = (const float*)d_in[0];
  const float* x_mark = (const float*)d_in[1];
  const float* emb_W  = (const float*)d_in[4];
  const float* emb_b  = (const float*)d_in[5];
  const float* mf[9], *mr[9];
  for (int i = 0; i < 9; ++i) { mf[i] = (const float*)d_in[6 + i]; mr[i] = (const float*)d_in[15 + i]; }
  const float* conv1_W = (const float*)d_in[24];
  const float* conv1_b = (const float*)d_in[25];
  const float* conv2_W = (const float*)d_in[26];
  const float* conv2_b = (const float*)d_in[27];
  const float* ln1_w = (const float*)d_in[28];
  const float* ln1_b = (const float*)d_in[29];
  const float* ln2_w = (const float*)d_in[30];
  const float* ln2_b = (const float*)d_in[31];
  const float* enc_w = (const float*)d_in[32];
  const float* enc_b = (const float*)d_in[33];
  const float* proj_W = (const float*)d_in[34];
  const float* proj_b = (const float*)d_in[35];

  // ---------------- workspace carve-up ----------------
  // fp32 region
  float* ws = (float*)d_ws;
  size_t o = 0;
  float* meanb = ws + o; o += (size_t)BB * NVARS;
  float* stdb  = ws + o; o += (size_t)BB * NVARS;
  float* x     = ws + o; o += (size_t)MROWS * DM;
  float* xsum  = ws + o; o += (size_t)MROWS * DM;
  float* xz    = ws + o; o += (size_t)MROWS * 2 * DM;  // head reused as dectmp
  float* xc    = ws + o; o += (size_t)MROWS * DM;
  float* dbc   = ws + o; o += (size_t)MROWS * 64;
  float* dtbuf = ws + o; o += (size_t)MROWS * DM;
  float* ybuf  = ws + o; o += (size_t)MROWS * DM;
  float* dirb  = ws + o; o += (size_t)MROWS * DM;
  // bf16 region (starts at 64B-aligned float offset)
  ushort_t* wsb = (ushort_t*)(ws + o);
  size_t ob = 0;
  ushort_t* tokb   = wsb + ob; ob += (size_t)MROWS * SEQ;
  ushort_t* x_bf   = wsb + ob; ob += (size_t)MROWS * DM;
  ushort_t* xrev_bf= wsb + ob; ob += (size_t)MROWS * DM;
  ushort_t* xc_bf  = wsb + ob; ob += (size_t)MROWS * DM;
  ushort_t* dbc_bf = wsb + ob; ob += (size_t)MROWS * 64;
  ushort_t* g_bf   = wsb + ob; ob += (size_t)MROWS * DM;  // also ffn1_bf
  // bf16 weights
  ushort_t* wemb  = wsb + ob; ob += (size_t)DM * SEQ;
  ushort_t* wproj = wsb + ob; ob += (size_t)PRED * DM;
  ushort_t* wc1   = wsb + ob; ob += (size_t)2 * DM * DM;
  ushort_t* wc2   = wsb + ob; ob += (size_t)2 * DM * DM;
  ushort_t* win[2], *wxp[2], *wdt[2], *wout[2];
  for (int d2 = 0; d2 < 2; ++d2) {
    win[d2]  = wsb + ob; ob += (size_t)2 * 2 * DM * DM;
    wxp[d2]  = wsb + ob; ob += (size_t)2 * 64 * DM;
    wdt[d2]  = wsb + ob; ob += (size_t)2 * DM * DTRANK;
    wout[d2] = wsb + ob; ob += (size_t)2 * DM * DM;
  }

  const size_t EW = (size_t)MROWS * DM;

  // ---------------- weight conversion (once per call) ----------------
  cvt_bf16_kernel<<<nb((size_t)DM * SEQ), 256, 0, stream>>>(emb_W, wemb, (size_t)DM * SEQ);
  cvt_bf16_kernel<<<nb((size_t)PRED * DM), 256, 0, stream>>>(proj_W, wproj, (size_t)PRED * DM);
  cvt_bf16_kernel<<<nb((size_t)2 * DM * DM), 256, 0, stream>>>(conv1_W, wc1, (size_t)2 * DM * DM);
  cvt_bf16_kernel<<<nb((size_t)2 * DM * DM), 256, 0, stream>>>(conv2_W, wc2, (size_t)2 * DM * DM);
  for (int d2 = 0; d2 < 2; ++d2) {
    const float* const* Wm = d2 ? mr : mf;
    cvt_bf16_kernel<<<nb((size_t)2 * 2 * DM * DM), 256, 0, stream>>>(Wm[0], win[d2], (size_t)2 * 2 * DM * DM);
    cvt_bf16_kernel<<<nb((size_t)2 * 64 * DM), 256, 0, stream>>>(Wm[3], wxp[d2], (size_t)2 * 64 * DM);
    cvt_bf16_kernel<<<nb((size_t)2 * DM * DTRANK), 256, 0, stream>>>(Wm[4], wdt[d2], (size_t)2 * DM * DTRANK);
    cvt_bf16_kernel<<<nb((size_t)2 * DM * DM), 256, 0, stream>>>(Wm[8], wout[d2], (size_t)2 * DM * DM);
  }

  // ---------------- forward ----------------
  prep_kernel<<<nb(BB * LTOK), 256, 0, stream>>>(x_enc, x_mark, tokb, meanb, stdb);
  // x = tok @ emb_W^T + emb_b   (f32 for residual, bf16 for in_proj GEMM)
  gemm_wmma_kernel<2, 4, 0, 3><<<dim3(MROWS / 32, DM / 64), 32, 0, stream>>>(
      tokb, SEQ, wemb, SEQ, emb_b, x, x_bf, DM, SEQ);

  for (int l = 0; l < 2; ++l) {
    flip_bf_kernel<<<nb(EW), 256, 0, stream>>>(x_bf, xrev_bf);
    for (int dir = 0; dir < 2; ++dir) {
      const float* const* Wm = dir ? mr : mf;
      const ushort_t* xin = dir ? xrev_bf : x_bf;
      const ushort_t* inW   = win[dir]  + (size_t)l * 2 * DM * DM;
      const ushort_t* xprojW= wxp[dir]  + (size_t)l * 64 * DM;
      const ushort_t* dtW   = wdt[dir]  + (size_t)l * DM * DTRANK;
      const ushort_t* outW  = wout[dir] + (size_t)l * DM * DM;
      const float* convW = Wm[1] + (size_t)l * DM * 2;
      const float* convB = Wm[2] + (size_t)l * DM;
      const float* dtB   = Wm[5] + (size_t)l * DM;
      const float* Alog  = Wm[6] + (size_t)l * DM * DSTATE;
      const float* Dvec  = Wm[7] + (size_t)l * DM;

      // xz = xin @ in_W^T
      gemm_wmma_kernel<2, 4, 0, 1><<<dim3(MROWS / 32, (2 * DM) / 64), 32, 0, stream>>>(
          xin, DM, inW, DM, nullptr, xz, nullptr, 2 * DM, DM);
      conv_silu_kernel<<<nb(EW), 256, 0, stream>>>(xz, convW, convB, xc, xc_bf);
      // dbc = xc @ xproj_W^T  (f32 for scan B/C, bf16 for dt GEMM)
      gemm_wmma_kernel<2, 2, 0, 3><<<dim3(MROWS / 32, 64 / 32), 32, 0, stream>>>(
          xc_bf, DM, xprojW, DM, nullptr, dbc, dbc_bf, 64, DM);
      // dt = softplus(dtr @ dt_W^T + dt_b)
      gemm_wmma_kernel<2, 4, 2, 1><<<dim3(MROWS / 32, DM / 64), 32, 0, stream>>>(
          dbc_bf, 64, dtW, DTRANK, dtB, dtbuf, nullptr, DM, DTRANK);
      scan_kernel<<<(BB * DM * DSTATE) / 256, 256, 0, stream>>>(
          xc, dtbuf, dbc, Alog, Dvec, ybuf);
      gate_kernel<<<nb(EW), 256, 0, stream>>>(ybuf, xz, g_bf);
      // dir output = g @ out_W^T
      gemm_wmma_kernel<2, 4, 0, 1><<<dim3(MROWS / 32, DM / 64), 32, 0, stream>>>(
          g_bf, DM, outW, DM, nullptr, dirb, nullptr, DM, DM);
      if (dir == 0)
        add_kernel<<<nb(EW), 256, 0, stream>>>(x, dirb, xsum);
      else
        combine_kernel<<<nb(EW), 256, 0, stream>>>(xsum, dirb, x);
    }
    // x = LN1(x)  (f32 + bf16 for FFN GEMM)
    ln_kernel<<<MROWS / 8, 256, 0, stream>>>(x, nullptr, ln1_w + l * DM, ln1_b + l * DM, x, x_bf);
    // FFN: ffn1 = relu(x@W1+b1) (bf16 only); ffn2 = ffn1@W2+b2 (f32)
    gemm_wmma_kernel<2, 4, 1, 2><<<dim3(MROWS / 32, DM / 64), 32, 0, stream>>>(
        x_bf, DM, wc1 + (size_t)l * DM * DM, DM, conv1_b + l * DM, nullptr, g_bf, DM, DM);
    gemm_wmma_kernel<2, 4, 0, 1><<<dim3(MROWS / 32, DM / 64), 32, 0, stream>>>(
        g_bf, DM, wc2 + (size_t)l * DM * DM, DM, conv2_b + l * DM, dirb, nullptr, DM, DM);
    // x = LN2(x + ffn_out)  (f32 + bf16 for next layer / final norm)
    ln_kernel<<<MROWS / 8, 256, 0, stream>>>(x, dirb, ln2_w + l * DM, ln2_b + l * DM, x, x_bf);
  }

  // encoder norm (bf16 feeds projection GEMM)
  ln_kernel<<<MROWS / 8, 256, 0, stream>>>(x, nullptr, enc_w, enc_b, nullptr, x_bf);
  float* dectmp = xz;  // reuse (free now)
  gemm_wmma_kernel<2, 2, 0, 1><<<dim3(MROWS / 32, PRED / 32), 32, 0, stream>>>(
      x_bf, DM, wproj, DM, proj_b, dectmp, nullptr, PRED, DM);
  denorm_kernel<<<nb((size_t)BB * PRED * NVARS), 256, 0, stream>>>(
      dectmp, meanb, stdb, (float*)d_out);
}
